// GeometricDataRotator_87522843561673
// MI455X (gfx1250) — matrix operand order repaired and checked by reading, not compiled, not run on hardware
//
#include <hip/hip_runtime.h>

typedef __attribute__((ext_vector_type(16))) __bf16 v16bf;
typedef __attribute__((ext_vector_type(8)))  float  v8f;

#define NE 8
#define BS 16384           // B*S
#define ND 512
#define NR 128
#define RM_PITCH 132
#define MAX_MAG 0.1f

#define WHI_OFF (1ull << 20)
#define WLO_OFF (WHI_OFF + (size_t)NE * ND * ND * 2)

// ---------------------------------------------------------------------------
// Kernel 1: build the 129x129 Givens-chain block per expert (closed form).
// All rotations are planes (0, j), j=1..128, applied sequentially.
// ---------------------------------------------------------------------------
__global__ void rot_block_kernel(const float* __restrict__ theta,
                                 float* __restrict__ rm_ws) {
  const int e = blockIdx.x;
  __shared__ float sc[NR], ss[NR], sp[NR];
  const int tid = threadIdx.x;
  if (tid < NR) {
    float a = tanhf(theta[e * NR + tid]) * MAX_MAG;
    sc[tid] = cosf(a);
    ss[tid] = sinf(a);
  }
  __syncthreads();
  if (tid == 0) {
    float p = 1.0f;
    for (int r = 0; r < NR; ++r) { p *= sc[r]; sp[r] = p; }
  }
  __syncthreads();

  float* rm = rm_ws + (size_t)e * 129 * RM_PITCH;
  const float Pall = sp[NR - 1];
  for (int idx = tid; idx < 129 * 129; idx += blockDim.x) {
    int k = idx / 129;          // row
    int d = idx % 129;          // col
    float val;
    if (d == 0) {
      // final column 0: [P_all ; s_r' * P_all/P_r']
      val = (k == 0) ? Pall : ss[k - 1] * (Pall / sp[k - 1]);
    } else {
      int r = d - 1;                                 // rotation that made col d
      float Pm1 = (r > 0) ? sp[r - 1] : 1.0f;        // prod c[0..r-1]
      if (k == 0) {
        val = -ss[r] * Pm1;
      } else {
        int rp = k - 1;
        if (rp < r)       val = -ss[r] * ss[rp] * (Pm1 / sp[rp]);
        else if (rp == r) val = sc[r];
        else              val = 0.0f;
      }
    }
    rm[(size_t)k * RM_PITCH + d] = val;
  }
}

// ---------------------------------------------------------------------------
// Kernel 2: W_e = proj_w[e] @ Rm[e] (only cols 0..128 differ), split to
// bf16 hi/lo pair for the bf16x3 WMMA GEMM.
// ---------------------------------------------------------------------------
__global__ void build_w_kernel(const float* __restrict__ proj_w,
                               const float* __restrict__ rm_ws,
                               unsigned short* __restrict__ whi,
                               unsigned short* __restrict__ wlo) {
  const int e = blockIdx.y;
  const int gid = blockIdx.x * blockDim.x + threadIdx.x;   // 0 .. 512*512-1
  const int o = gid >> 9;
  const int d = gid & 511;
  const float* pw = proj_w + (size_t)e * ND * ND + (size_t)o * ND;
  float w;
  if (d >= 129) {
    w = pw[d];
  } else {
    const float* rm = rm_ws + (size_t)e * 129 * RM_PITCH;
    float acc = 0.0f;
    for (int k = 0; k < 129; ++k)
      acc = fmaf(pw[k], rm[(size_t)k * RM_PITCH + d], acc);
    w = acc;
  }
  __bf16 h = (__bf16)w;
  __bf16 l = (__bf16)(w - (float)h);
  size_t off = (size_t)e * ND * ND + (size_t)gid;
  whi[off] = __builtin_bit_cast(unsigned short, h);
  wlo[off] = __builtin_bit_cast(unsigned short, l);
}

// ---------------------------------------------------------------------------
// Kernel 3: per-expert GEMM out[e] = X @ W_e^T with bf16x3 split-precision
// WMMA. Workgroup tile 256(M) x 64(N), 8 waves of 64x32, K step 32.
// ---------------------------------------------------------------------------
__global__ void __launch_bounds__(256)
gemm_kernel(const float* __restrict__ x,
            const unsigned short* __restrict__ whi,
            const unsigned short* __restrict__ wlo,
            float* __restrict__ out) {
  const int e    = blockIdx.z;
  const int n0   = blockIdx.x * 64;
  const int m0   = blockIdx.y * 256;
  const int tid  = threadIdx.x;
  const int lane = tid & 31;
  const int wave = tid >> 5;
  const int wm   = wave >> 1;        // 0..3
  const int wn   = wave & 1;         // 0..1
  const int mwave = m0 + wm * 64;
  const int nwave = n0 + wn * 32;
  const int lhalf = lane >> 4;       // 0|1
  const int l16   = lane & 15;

  // B tiles staged in LDS: [hi/lo][64 rows][32 k], pitch 40 ushorts (80B)
  __shared__ unsigned short sB[2][64 * 40];

  const unsigned short* Wh = whi + (size_t)e * ND * ND;
  const unsigned short* Wl = wlo + (size_t)e * ND * ND;

  const v8f zacc = {0.f, 0.f, 0.f, 0.f, 0.f, 0.f, 0.f, 0.f};
  v8f acc[4][2];
#pragma unroll
  for (int i = 0; i < 4; ++i)
#pragma unroll
    for (int j = 0; j < 2; ++j) acc[i][j] = zacc;

  union BF { v16bf v; uint4 q[2]; };

  const int srow = tid >> 2;   // 0..63 staging row
  const int sc4  = tid & 3;    // 0..3 staging 16B chunk

  for (int k0 = 0; k0 < ND; k0 += 32) {
    __syncthreads();
    // ---- stage W hi/lo tile (64 rows x 32 k of bf16) into LDS ----
    {
      const uint4* gh = (const uint4*)(Wh + (size_t)(n0 + srow) * ND + k0 + sc4 * 8);
      *(uint4*)(&sB[0][srow * 40 + sc4 * 8]) = *gh;
      const uint4* gl = (const uint4*)(Wl + (size_t)(n0 + srow) * ND + k0 + sc4 * 8);
      *(uint4*)(&sB[1][srow * 40 + sc4 * 8]) = *gl;
    }
    __syncthreads();

    // ---- B fragments (32x16 bf16): lane<16 -> N=lane,K 0..15; lane>=16 -> K 16..31
    BF bhi[2], blo[2];
#pragma unroll
    for (int nt = 0; nt < 2; ++nt) {
      int nl = wn * 32 + nt * 16 + l16;
      int ko = lhalf * 16;
      const uint4* ph = (const uint4*)(&sB[0][nl * 40 + ko]);
      bhi[nt].q[0] = ph[0]; bhi[nt].q[1] = ph[1];
      const uint4* pl = (const uint4*)(&sB[1][nl * 40 + ko]);
      blo[nt].q[0] = pl[0]; blo[nt].q[1] = pl[1];
    }

    // ---- A fragments: load fp32 X rows, split to bf16 hi/lo on the fly ----
    // lane<16: M=lane, K 0..15 ; lane>=16: M=lane-16, K 16..31
    BF ahi[4], alo[4];
#pragma unroll
    for (int mt = 0; mt < 4; ++mt) {
      int m   = mwave + mt * 16 + l16;
      int kb1 = k0 + lhalf * 16;
      const float4* p1 = (const float4*)(x + (size_t)m * ND + kb1);
      const float4* p2 = (const float4*)(x + (size_t)m * ND + kb1 + 8);
      float4 a0 = p1[0], a1 = p1[1], b0 = p2[0], b1 = p2[1];
      float f[16] = {a0.x, a0.y, a0.z, a0.w, a1.x, a1.y, a1.z, a1.w,
                     b0.x, b0.y, b0.z, b0.w, b1.x, b1.y, b1.z, b1.w};
#pragma unroll
      for (int i = 0; i < 16; ++i) {
        __bf16 h = (__bf16)f[i];
        ahi[mt].v[i] = h;
        alo[mt].v[i] = (__bf16)(f[i] - (float)h);
      }
    }

    // ---- 24 WMMAs: hi*hi + hi*lo + lo*hi ----
#pragma unroll
    for (int mt = 0; mt < 4; ++mt)
#pragma unroll
      for (int nt = 0; nt < 2; ++nt) {
        acc[mt][nt] = __builtin_amdgcn_wmma_f32_16x16x32_bf16(
            false, ahi[mt].v, false, bhi[nt].v, (short)0, acc[mt][nt], false, false);
        acc[mt][nt] = __builtin_amdgcn_wmma_f32_16x16x32_bf16(
            false, ahi[mt].v, false, blo[nt].v, (short)0, acc[mt][nt], false, false);
        acc[mt][nt] = __builtin_amdgcn_wmma_f32_16x16x32_bf16(
            false, alo[mt].v, false, bhi[nt].v, (short)0, acc[mt][nt], false, false);
      }
  }

  // ---- epilogue: C/D layout -> out[e][m][n] ----
  float* O = out + (size_t)e * BS * ND;
#pragma unroll
  for (int mt = 0; mt < 4; ++mt)
#pragma unroll
    for (int nt = 0; nt < 2; ++nt) {
      int col = nwave + nt * 16 + l16;
#pragma unroll
      for (int v = 0; v < 8; ++v) {
        int row = mwave + mt * 16 + v + 8 * lhalf;
        O[(size_t)row * ND + col] = acc[mt][nt][v];
      }
    }
}

// ---------------------------------------------------------------------------
extern "C" void kernel_launch(void* const* d_in, const int* in_sizes, int n_in,
                              void* d_out, int out_size, void* d_ws, size_t ws_size,
                              hipStream_t stream) {
  const float* x      = (const float*)d_in[0];   // (8, 2048, 512)
  const float* theta  = (const float*)d_in[1];   // (8, 128)
  const float* proj_w = (const float*)d_in[2];   // (8, 512, 512)
  float* out = (float*)d_out;                    // (8, 8, 2048, 512)

  float*          rm_ws = (float*)d_ws;                               // 8*129*132*4 B
  unsigned short* whi   = (unsigned short*)((char*)d_ws + WHI_OFF);   // 4 MiB
  unsigned short* wlo   = (unsigned short*)((char*)d_ws + WLO_OFF);   // 4 MiB

  rot_block_kernel<<<dim3(NE), dim3(256), 0, stream>>>(theta, rm_ws);
  build_w_kernel<<<dim3((ND * ND) / 256, NE), dim3(256), 0, stream>>>(proj_w, rm_ws, whi, wlo);
  gemm_kernel<<<dim3(ND / 64, BS / 256, NE), dim3(256), 0, stream>>>(x, whi, wlo, out);
}
